// EXL2Linear_31095563223122
// MI455X (gfx1250) — matrix operand order repaired and checked by reading, not compiled, not run on hardware
//
#include <hip/hip_runtime.h>
#include <hip/hip_bf16.h>

// ---------------------------------------------------------------------------
// EXL2Linear forward, MI455X (gfx1250, wave32, WMMA).
//
// Reference quirk: only input columns 0..63 carry unpacked 4-bit codes; all
// other columns of the dequantized W reduce to the per-group zero point.
//   out[m,o] = qscale[o,0] * sum_{c<64} x[m,c]*U[o,c]
//            + sum_{g<32}  qzero[o,g] * groupsum_g(x[m,:])
//            + bias[o]
// => M=8192, N=4096, K=96 GEMM with f16 panels / f32 accumulate via
//    v_wmma_f32_16x16x32_f16. Memory-bound: one pass over x, one write of out.
// ---------------------------------------------------------------------------

typedef __attribute__((ext_vector_type(16))) _Float16 v16h;
typedef __attribute__((ext_vector_type(8)))  _Float16 v8h;
typedef __attribute__((ext_vector_type(8)))  float    v8f;

#define I_DIM 4096
#define O_DIM 4096
#define KP    96        // 64 code columns + 32 group sums
#define GROUPS 32

// ---------------------------------------------------------------------------
// Kernel 1: A panel. One block per row of x.
//   A[m, 0:64]  = (f16) x[m, 0:64]
//   A[m, 64+g]  = (f16) sum of x[m, g*128 : g*128+128]
// ---------------------------------------------------------------------------
__global__ __launch_bounds__(256) void exl2_prep_a(const float* __restrict__ x,
                                                   _Float16* __restrict__ A) {
    const int row = blockIdx.x;
    const int t   = threadIdx.x;              // 256 threads
    const float* xr = x + (size_t)row * I_DIM;

    // Thread t sums 16 consecutive floats (4 x float4); 4 loads tile 2 KB/wave.
    const float4* xr4 = (const float4*)xr;
    float s = 0.0f;
#pragma unroll
    for (int i = 0; i < 4; ++i) {
        float4 v = xr4[t * 4 + i];
        s += v.x + v.y + v.z + v.w;
    }

    __shared__ float part[256];
    part[t] = s;
    __syncthreads();

    _Float16* ar = A + (size_t)row * KP;
    if (t < 64) {
        ar[t] = (_Float16)xr[t];              // head columns, exact role in GEMM
    }
    if (t < GROUPS) {                         // 8 partials (of 16 floats) per group
        float g = 0.0f;
#pragma unroll
        for (int i = 0; i < 8; ++i) g += part[t * 8 + i];
        ar[64 + t] = (_Float16)g;
    }
}

// ---------------------------------------------------------------------------
// Kernel 2: B panel (dequant folded).  B[o, j]:
//   j < 64 : qscale[o,0] * ((qweight[o, j%8] >> 4*(j/8)) & 15)
//   j >= 64: qzero[o, j-64]
// ---------------------------------------------------------------------------
__global__ __launch_bounds__(256) void exl2_prep_b(const int*   __restrict__ qweight,
                                                   const float* __restrict__ qscale,
                                                   const int*   __restrict__ qzero,
                                                   _Float16*    __restrict__ B) {
    const int idx = blockIdx.x * 256 + threadIdx.x;   // O_DIM * KP total
    if (idx >= O_DIM * KP) return;
    const int o = idx / KP;
    const int j = idx - o * KP;
    float v;
    if (j < 64) {
        const int word = qweight[(size_t)o * (I_DIM / 8) + (j & 7)];
        const int u    = (word >> ((j >> 3) * 4)) & 15;
        v = (float)u * qscale[(size_t)o * GROUPS];
    } else {
        v = (float)qzero[(size_t)o * GROUPS + (j - 64)];
    }
    B[idx] = (_Float16)v;
}

// ---------------------------------------------------------------------------
// Fragment loaders (wave32; contiguous k order, same for A and B).
// A 16x32 f16: lane L -> row M = L&15, kbase = 16*(L>>4);
//   elements 0..15 = K kbase+0..15 (two 8-element loads).
// B 32x16 f16: lane L -> col N = L&15, kbase = 16*(L>>4);
//   elements 0..15 = K kbase+0..15 (sequential).
// ---------------------------------------------------------------------------
__device__ __forceinline__ v16h load_a_frag(const _Float16* rowbase, int lane) {
    const int kh = (lane >> 4) * 16;
    v8h lo = *(const v8h*)(rowbase + kh);
    v8h hi = *(const v8h*)(rowbase + kh + 8);
    v16h f;
#pragma unroll
    for (int i = 0; i < 8; ++i) { f[i] = lo[i]; f[i + 8] = hi[i]; }
    return f;
}

__device__ __forceinline__ v16h load_b_frag(const _Float16* rowbase, int lane) {
    const int kb = (lane >> 4) * 16;
    return *(const v16h*)(rowbase + kb);      // 32B, 32B-aligned (ld=192B)
}

// ---------------------------------------------------------------------------
// Kernel 3: WMMA GEMM  out[m,o] = A[m,:] . B[o,:] + bias[o]
// Block = 128 threads = 4 waves; each wave owns a 16(M) x 64(N) tile.
// grid.x = N/64, grid.y = M/64.
// ---------------------------------------------------------------------------
__global__ __launch_bounds__(128) void exl2_wmma_gemm(const _Float16* __restrict__ A,
                                                      const _Float16* __restrict__ B,
                                                      const float*    __restrict__ bias,
                                                      float*          __restrict__ out) {
    const int lane   = threadIdx.x & 31;
    const int waveId = threadIdx.x >> 5;
    const int m0 = (blockIdx.y * 4 + waveId) * 16;
    const int n0 = blockIdx.x * 64;

    v8f acc[4] = {};

    const _Float16* aRow = A + (size_t)(m0 + (lane & 15)) * KP;

#pragma unroll
    for (int kt = 0; kt < 3; ++kt) {          // K = 96 = 3 x 32
        const v16h a = load_a_frag(aRow + kt * 32, lane);
#pragma unroll
        for (int j = 0; j < 4; ++j) {
            const _Float16* bRow = B + (size_t)(n0 + j * 16 + (lane & 15)) * KP + kt * 32;
            const v16h b = load_b_frag(bRow, lane);
            acc[j] = __builtin_amdgcn_wmma_f32_16x16x32_f16(
                false, a, false, b, (short)0, acc[j], false, false);
        }
    }

    // C/D layout: VGPR v -> M = m0 + v + 8*(lane>>4), N = n0 + (lane&15).
    const int col  = lane & 15;
    const int rsel = lane >> 4;
#pragma unroll
    for (int j = 0; j < 4; ++j) {
        const int n  = n0 + j * 16 + col;
        const float bv = bias[n];
#pragma unroll
        for (int v = 0; v < 8; ++v) {
            const int m = m0 + v + rsel * 8;
            out[(size_t)m * O_DIM + n] = acc[j][v] + bv;
        }
    }
}

// ---------------------------------------------------------------------------
extern "C" void kernel_launch(void* const* d_in, const int* in_sizes, int n_in,
                              void* d_out, int out_size, void* d_ws, size_t ws_size,
                              hipStream_t stream) {
    const float* x       = (const float*)d_in[0];   // [M, 4096]
    const int*   qweight = (const int*)  d_in[1];   // [4096, 512]
    const float* qscale  = (const float*)d_in[2];   // [4096, 32]
    const int*   qzero   = (const int*)  d_in[3];   // [4096, 32]
    const float* bias    = (const float*)d_in[4];   // [4096]
    float*       out     = (float*)d_out;

    const int M = in_sizes[0] / I_DIM;              // 4*2048 = 8192 rows

    _Float16* A = (_Float16*)d_ws;                  // [M, 96] f16
    _Float16* B = A + (size_t)M * KP;               // [4096, 96] f16

    exl2_prep_a<<<M, 256, 0, stream>>>(x, A);

    exl2_prep_b<<<(O_DIM * KP + 255) / 256, 256, 0, stream>>>(qweight, qscale, qzero, B);

    dim3 grid(O_DIM / 64, M / 64);                  // (64, 128)
    exl2_wmma_gemm<<<grid, 128, 0, stream>>>(A, B, bias, out);
}
